// CREStereo_12833362280738
// MI455X (gfx1250) — matrix operand order repaired and checked by reading, not compiled, not run on hardware
//
#include <hip/hip_runtime.h>

typedef __attribute__((ext_vector_type(16))) _Float16     v16h;
typedef __attribute__((ext_vector_type(8)))  float        v8f;
typedef __attribute__((ext_vector_type(8)))  unsigned int v8u;

#define Bn 2
#define Cn 256
#define Hn 96
#define Wn 192
#define Gn 4
#define Kn 9
#define HWn (Hn * Wn)

// One wave32 per pixel.
// D[k,g] = sum_c S[k,c] * (left[c] masked to group g), contracted over 256
// channels in 8 chunks of K=32 via v_wmma_f32_16x16x32_f16.
//  A (16x32 f16): lane L (row k=L%16), VGPR v halves hold channels
//     c = 32t + 2v + (L>=16 ? 16 : 0) + half
//  B (32x16 f16): lane n (col g=n%16), vector idx i holds channel
//     c = 32t + i + (n>=16 ? 16 : 0); nonzero only when n == t>>1.
//     => lane n only ever needs the 32 channels of group n: hoisted out of loop.
//  D (16x16 f32): VGPR v: lanes 0-15 -> (k=v, g=lane); lanes 16-31 -> (k=v+8, g=lane-16)
__global__ __launch_bounds__(256) void crestereo_corr_wmma(
    const float* __restrict__ left, const float* __restrict__ right,
    const float* __restrict__ flow, const float* __restrict__ extra,
    float* __restrict__ out)
{
  const int lane  = threadIdx.x & 31;
  const int pixel = blockIdx.x * 8 + (threadIdx.x >> 5);
  if (pixel >= Bn * HWn) return;            // wave-uniform

  const int b  = pixel / HWn;
  const int hw = pixel - b * HWn;
  const int h  = hw / Wn;
  const int w  = hw - h * Wn;

  const int  col    = lane & 15;            // A: row k ; B/D: column g
  const bool hiLane = lane >= 16;
  const int  k      = col;
  const bool kValid = (k < Kn);
  const int  kk     = (k < Kn - 1) ? k : (Kn - 1);   // clamped for addressing

  // flow at this pixel (wave-uniform loads)
  const float fx = flow[(b * 2 + 0) * HWn + hw];
  const float fy = flow[(b * 2 + 1) * HWn + hw];

  // per-lane (= per-candidate) bilinear corner weights/offsets, fully branchless:
  // invalid-k lanes get zero weights and safe in-bounds offsets.
  const float ex = extra[((b * (2 * Kn)) + 2 * kk + 0) * HWn + hw];
  const float ey = extra[((b * (2 * Kn)) + 2 * kk + 1) * HWn + hw];
  const float cx = (float)w + fx + (float)(kk - 4) + ex;
  const float cy = (float)h + fy + ey;
  const float fx0 = floorf(cx), fy0 = floorf(cy);
  const float wx1 = cx - fx0,   wy1 = cy - fy0;
  const float wx0 = 1.f - wx1,  wy0 = 1.f - wy1;
  const int ix0 = (int)fx0, iy0 = (int)fy0;
  const int ix1 = ix0 + 1,  iy1 = iy0 + 1;
  const bool vx0 = (ix0 >= 0) && (ix0 < Wn);
  const bool vx1 = (ix1 >= 0) && (ix1 < Wn);
  const bool vy0 = (iy0 >= 0) && (iy0 < Hn);
  const bool vy1 = (iy1 >= 0) && (iy1 < Hn);
  const int ix0c = min(max(ix0, 0), Wn - 1), ix1c = min(max(ix1, 0), Wn - 1);
  const int iy0c = min(max(iy0, 0), Hn - 1), iy1c = min(max(iy1, 0), Hn - 1);
  const float kmask = kValid ? 1.f : 0.f;
  const float w00 = (vx0 && vy0) ? wx0 * wy0 * kmask : 0.f;
  const float w01 = (vx1 && vy0) ? wx1 * wy0 * kmask : 0.f;
  const float w10 = (vx0 && vy1) ? wx0 * wy1 * kmask : 0.f;
  const float w11 = (vx1 && vy1) ? wx1 * wy1 * kmask : 0.f;
  const int off00 = iy0c * Wn + ix0c, off01 = iy0c * Wn + ix1c;
  const int off10 = iy1c * Wn + ix0c, off11 = iy1c * Wn + ix1c;

  const float* rb = right + (size_t)b * Cn * HWn;       // right[b,:,:,:]
  const float* lb = left  + (size_t)b * Cn * HWn + hw;  // left[b,:,h,w] (stride HWn)

  // ---- B: hoisted. Lane (col,hi) needs left channels of group col only:
  //   chunk 2*col   -> channels 64*col      + (hi?16:0) + 0..15
  //   chunk 2*col+1 -> channels 64*col + 32 + (hi?16:0) + 0..15
  // Lanes with col>=4 load group 3 (in-bounds) and are masked off per chunk.
  const int gcol = (col < Gn) ? col : (Gn - 1);
  const float* plg = lb + (size_t)(64 * gcol + (hiLane ? 16 : 0)) * HWn;
  v16h bm0, bm1;
  #pragma unroll
  for (int i = 0; i < 16; ++i) {
    bm0[i] = (_Float16)plg[(size_t)i * HWn];
    bm1[i] = (_Float16)plg[(size_t)(32 + i) * HWn];
  }
  const v8u b0u = __builtin_bit_cast(v8u, bm0);
  const v8u b1u = __builtin_bit_cast(v8u, bm1);

  // Per-lane corner base pointers with the hi-lane channel shift folded in.
  // Advanced by 32*HWn per chunk so every in-chunk load offset is an immediate.
  const size_t hiA = (size_t)(hiLane ? 16 : 0) * HWn;
  const float* p00 = rb + hiA + off00;
  const float* p01 = rb + hiA + off01;
  const float* p10 = rb + hiA + off10;
  const float* p11 = rb + hiA + off11;

  v8f acc = {};

  #pragma unroll
  for (int t = 0; t < 8; ++t) {                 // channels [32t, 32t+32), group t>>1
    // ---- A: bilinear-sampled right feature for candidate k (branchless) ----
    v16h a;
    #pragma unroll
    for (int v = 0; v < 8; ++v) {
      const int cbase = 2 * v;
      #pragma unroll
      for (int hh = 0; hh < 2; ++hh) {
        const int co = (cbase + hh) * HWn;      // compile-time after unroll
        const float s = w00 * p00[co] + w01 * p01[co]
                      + w10 * p10[co] + w11 * p11[co];
        a[2 * v + hh] = (_Float16)s;
      }
    }
    // ---- B: select this lane's preloaded group channels, mask other columns ----
    const bool bAct = (col == (t >> 1));
    const v8u  bsrc = (t & 1) ? b1u : b0u;      // compile-time select after unroll
    v8u bsel;
    #pragma unroll
    for (int j = 0; j < 8; ++j) bsel[j] = bAct ? bsrc[j] : 0u;
    const v16h bmt = __builtin_bit_cast(v16h, bsel);

    acc = __builtin_amdgcn_wmma_f32_16x16x32_f16(
        /*neg_a=*/false, a, /*neg_b=*/false, bmt,
        /*c_mod=*/(short)0, acc, /*reuse_a=*/false, /*reuse_b=*/false);

    p00 += 32 * HWn; p01 += 32 * HWn; p10 += 32 * HWn; p11 += 32 * HWn;
  }

  // scatter D[k,g]/64 -> out[b, g*9+k, h, w]
  const float inv = 1.0f / 64.0f;
  if (col < Gn) {
    if (!hiLane) {
      #pragma unroll
      for (int v = 0; v < 8; ++v)   // k = v, g = col
        out[(size_t)(b * (Gn * Kn) + col * Kn + v) * HWn + hw] = acc[v] * inv;
    } else {                        // k = 8, g = col
      out[(size_t)(b * (Gn * Kn) + col * Kn + 8) * HWn + hw] = acc[0] * inv;
    }
  }
}

extern "C" void kernel_launch(void* const* d_in, const int* in_sizes, int n_in,
                              void* d_out, int out_size, void* d_ws, size_t ws_size,
                              hipStream_t stream) {
  const float* left  = (const float*)d_in[0];
  const float* right = (const float*)d_in[1];
  const float* flow  = (const float*)d_in[2];
  const float* extra = (const float*)d_in[3];
  float* out = (float*)d_out;

  const int totalWaves = Bn * Hn * Wn;            // one wave per pixel = 36864
  const int blocks = (totalWaves + 7) / 8;        // 8 waves (256 threads) per block
  crestereo_corr_wmma<<<blocks, 256, 0, stream>>>(left, right, flow, extra, out);
}
